// DSFMarginal_30846455120614
// MI455X (gfx1250) — matrix operand order repaired and checked by reading, not compiled, not run on hardware
//
#include <hip/hip_runtime.h>
#include <hip/hip_bf16.h>
#include <stdint.h>

// Problem constants (from reference)
#define BN_TOTAL (32 * 4096)   // B*N rows
#define CDIM 256
#define MLPD 256
#define PDIM 144               // FLOW_LAYERS*3*FLOW_HID = 3*3*16
#define KT   8                 // K tiles of 32 (K = 256 for all three GEMMs)
#define NT0  16                // 256 / 16
#define NT2  9                 // 144 / 16
#define MAX_ITER 100
#define WAVES_PER_BLOCK 4
#define ROWS_PER_WAVE   32     // two 16-row M-tiles per wave (B-fragment reuse x2)
#define ROWS_PER_BLOCK  (WAVES_PER_BLOCK * ROWS_PER_WAVE)  // 128

typedef __attribute__((ext_vector_type(16))) __bf16 v16bf;
typedef __attribute__((ext_vector_type(8)))  float  v8f;

union FragBF {                 // one WMMA 16x16x32 bf16 operand (8 VGPRs / lane)
  v16bf          bf;
  uint4          q[2];
  unsigned short us[16];
};

__device__ __forceinline__ unsigned short f2bf(float f) {
  // round-to-nearest-even fp32 -> bf16
  unsigned int x = __float_as_uint(f);
  unsigned int r = x + 0x7FFFu + ((x >> 16) & 1u);
  return (unsigned short)(r >> 16);
}

__device__ __forceinline__ float relu1(float x) {
  // v_med3_f32: clamp to [0, +inf) in a single VALU op
  return __builtin_amdgcn_fmed3f(x, 0.0f, __builtin_inff());
}

__device__ __forceinline__ void packchunk(FragBF& f, int o, float4 a, float4 b) {
  f.us[o + 0] = f2bf(a.x); f.us[o + 1] = f2bf(a.y);
  f.us[o + 2] = f2bf(a.z); f.us[o + 3] = f2bf(a.w);
  f.us[o + 4] = f2bf(b.x); f.us[o + 5] = f2bf(b.y);
  f.us[o + 6] = f2bf(b.z); f.us[o + 7] = f2bf(b.w);
}

// ---------------------------------------------------------------------------
// Kernel 1: fp32 weight [K=256][N] row-major -> bf16 in WMMA B-fragment order
// (ISA 7.12.2, 16-bit B 32x16): per fragment (kt,nt), lane L holds column
// n = nt*16 + L%16, K = kt*32 + (L/16)*8 + {0..7,16..23} as 8 pairs -> 16 bf16/lane.
// ---------------------------------------------------------------------------
__global__ void prep_weights(const float* __restrict__ W,
                             unsigned short* __restrict__ dst,
                             int nTiles, int N) {
  int tid = blockIdx.x * blockDim.x + threadIdx.x;
  int total = KT * nTiles * 32;
  if (tid >= total) return;
  int L    = tid & 31;
  int frag = tid >> 5;
  int kt   = frag / nTiles;
  int nt   = frag % nTiles;
  int g    = L >> 4;
  int n    = nt * 16 + (L & 15);
  unsigned short* o = dst + (size_t)tid * 16;
#pragma unroll
  for (int v = 0; v < 8; ++v) {
    int k = kt * 32 + g * 8 + 2 * v + (v >= 4 ? 8 : 0);
    o[2 * v + 0] = f2bf(W[(size_t)(k + 0) * N + n]);
    o[2 * v + 1] = f2bf(W[(size_t)(k + 1) * N + n]);
  }
}

// ---------------------------------------------------------------------------
// Kernel 2: fused 3-layer conditioner MLP.
// 4 waves/block, 32 rows (2 M-tiles)/wave: each B fragment loaded once per
// wave feeds 2 WMMAs. Context read once; h0/h1 bounce through a wave-private
// LDS slice (bf16, A-fragment-friendly row-major layout).
// ---------------------------------------------------------------------------
__device__ __forceinline__ void load_a_from_lds2(FragBF (*aFr)[KT],
                                                 const unsigned short* hw,
                                                 int g, int r16) {
#pragma unroll
  for (int t = 0; t < 2; ++t) {
    const unsigned short* row = hw + (t * 16 + r16) * MLPD;
#pragma unroll
    for (int kt = 0; kt < KT; ++kt) {
      // 16-bit A 16x32 layout: group 0 -> K {0..7,16..23}, group 1 -> {8..15,24..31}
      int base0 = kt * 32 + g * 8;
      aFr[t][kt].q[0] = *(const uint4*)(row + base0);
      aFr[t][kt].q[1] = *(const uint4*)(row + base0 + 16);
    }
  }
}

__device__ __forceinline__ void layer_to_lds2(const FragBF (*aFr)[KT],
                                              const unsigned short* __restrict__ wf,
                                              const float* __restrict__ bias,
                                              unsigned short* hw,
                                              int lane, int g, int r16) {
#pragma unroll 1
  for (int nt = 0; nt < NT0; ++nt) {
    int n = nt * 16 + r16;
    float bv = bias[n];
    v8f acc0, acc1;
#pragma unroll
    for (int r = 0; r < 8; ++r) { acc0[r] = bv; acc1[r] = bv; }
#pragma unroll
    for (int kt = 0; kt < KT; ++kt) {
      FragBF bFr;
      const uint4* src =
          (const uint4*)(wf + ((size_t)(kt * NT0 + nt) * 32 + lane) * 16);
      bFr.q[0] = src[0];
      bFr.q[1] = src[1];
      acc0 = __builtin_amdgcn_wmma_f32_16x16x32_bf16(
          false, aFr[0][kt].bf, false, bFr.bf, (short)0, acc0, false, false);
      acc1 = __builtin_amdgcn_wmma_f32_16x16x32_bf16(
          false, aFr[1][kt].bf, false, bFr.bf, (short)0, acc1, false, false);
    }
    // C/D layout: VGPR r -> M = (lane<16 ? r : 8+r), N = lane%16 of the tile
#pragma unroll
    for (int r = 0; r < 8; ++r) {
      int M = g * 8 + r;
      hw[(M)*MLPD + n]      = f2bf(relu1(acc0[r]));
      hw[(16 + M)*MLPD + n] = f2bf(relu1(acc1[r]));
    }
  }
}

__global__ __launch_bounds__(WAVES_PER_BLOCK * 32)
void conditioner(const float* __restrict__ ctx,
                 const unsigned short* __restrict__ w0f,
                 const unsigned short* __restrict__ w1f,
                 const unsigned short* __restrict__ w2f,
                 const float* __restrict__ b0,
                 const float* __restrict__ b1,
                 const float* __restrict__ b2,
                 float* __restrict__ params) {
  // 64 KB: 4 wave-private 32x256 bf16 slices
  __shared__ unsigned short hbuf[WAVES_PER_BLOCK * ROWS_PER_WAVE * MLPD];
  const int lane = threadIdx.x & 31;
  const int wave = threadIdx.x >> 5;
  const int g    = lane >> 4;
  const int r16  = lane & 15;
  const size_t m0 =
      ((size_t)blockIdx.x * WAVES_PER_BLOCK + wave) * ROWS_PER_WAVE;
  unsigned short* hw = hbuf + wave * (ROWS_PER_WAVE * MLPD);

  // Prefetch next block's context tile into the cache hierarchy.
  if ((int)blockIdx.x + 1 < (int)gridDim.x) {
    const float* nf = ctx + ((size_t)(blockIdx.x + 1) * ROWS_PER_BLOCK) * CDIM +
                      (size_t)threadIdx.x * 256;
    __builtin_prefetch(nf, 0, 1);
  }

  // ---- A fragments for layer 0 straight from fp32 context ----
  FragBF aFr[2][KT];
#pragma unroll
  for (int t = 0; t < 2; ++t) {
    const float* row = ctx + (m0 + (size_t)(t * 16 + r16)) * CDIM;
#pragma unroll
    for (int kt = 0; kt < KT; ++kt) {
      int base0 = kt * 32 + g * 8;
      float4 c0 = *(const float4*)(row + base0);
      float4 c1 = *(const float4*)(row + base0 + 4);
      float4 c2 = *(const float4*)(row + base0 + 16);
      float4 c3 = *(const float4*)(row + base0 + 20);
      packchunk(aFr[t][kt], 0, c0, c1);
      packchunk(aFr[t][kt], 8, c2, c3);
    }
  }

  // Layer 0: h0 = relu(ctx @ W0 + b0) -> LDS (bf16)
  layer_to_lds2(aFr, w0f, b0, hw, lane, g, r16);
  asm volatile("s_wait_dscnt 0" ::: "memory");

  // Layer 1: h1 = relu(h0 @ W1 + b1) -> LDS (bf16)
  load_a_from_lds2(aFr, hw, g, r16);
  asm volatile("s_wait_dscnt 0" ::: "memory");
  layer_to_lds2(aFr, w1f, b1, hw, lane, g, r16);
  asm volatile("s_wait_dscnt 0" ::: "memory");

  // Layer 2: params = h1 @ W2 + b2 -> global fp32
  load_a_from_lds2(aFr, hw, g, r16);
#pragma unroll 1
  for (int nt = 0; nt < NT2; ++nt) {
    int n = nt * 16 + r16;
    float bv = b2[n];
    v8f acc0, acc1;
#pragma unroll
    for (int r = 0; r < 8; ++r) { acc0[r] = bv; acc1[r] = bv; }
#pragma unroll
    for (int kt = 0; kt < KT; ++kt) {
      FragBF bFr;
      const uint4* src =
          (const uint4*)(w2f + ((size_t)(kt * NT2 + nt) * 32 + lane) * 16);
      bFr.q[0] = src[0];
      bFr.q[1] = src[1];
      acc0 = __builtin_amdgcn_wmma_f32_16x16x32_bf16(
          false, aFr[0][kt].bf, false, bFr.bf, (short)0, acc0, false, false);
      acc1 = __builtin_amdgcn_wmma_f32_16x16x32_bf16(
          false, aFr[1][kt].bf, false, bFr.bf, (short)0, acc1, false, false);
    }
#pragma unroll
    for (int r = 0; r < 8; ++r) {
      params[(m0 + (size_t)(g * 8 + r)) * PDIM + n]      = acc0[r];
      params[(m0 + (size_t)(16 + g * 8 + r)) * PDIM + n] = acc1[r];
    }
  }
}

// ---------------------------------------------------------------------------
// Kernel 3: per-element bisection on the Deep Sigmoid Flow CDF.
// All base-e <-> base-2 constants folded into the iteration-invariant
// precompute: layer 0 uses A*log2e, B*log2e; the inter-layer logit is kept in
// log2 units (log2(xp)-log2(1-xp)), which cancels the log2e for layers 1/2.
// Inner sigmoid = fma, v_exp_f32, add, v_rcp_f32, fma — no stray multiplies.
// ---------------------------------------------------------------------------
__global__ __launch_bounds__(256)
void bisect(const float* __restrict__ params, const float* __restrict__ u,
            float* __restrict__ out) {
  const float LOG2E = 1.4426950408889634f;
  int idx = blockIdx.x * 256 + threadIdx.x;
  const float* p = params + (size_t)idx * PDIM;

  float pr[PDIM];
#pragma unroll
  for (int i = 0; i < PDIM / 4; ++i) {
    float4 v = *(const float4*)(p + 4 * i);
    pr[4 * i + 0] = v.x; pr[4 * i + 1] = v.y;
    pr[4 * i + 2] = v.z; pr[4 * i + 3] = v.w;
  }

  float A[48], Bb[48], Wt[48];
#pragma unroll
  for (int l = 0; l < 3; ++l) {
    int off = l * 48;
    float sc = (l == 0) ? LOG2E : 1.0f;  // layers 1/2 consume log2-unit inputs
#pragma unroll
    for (int j = 0; j < 16; ++j) {
      float x = pr[off + j];
      // numerically-stable softplus (one-time cost)
      float sp = fmaxf(x, 0.0f) + __logf(1.0f + __expf(-fabsf(x)));
      A[l * 16 + j]  = sp * sc;
      Bb[l * 16 + j] = pr[off + 16 + j] * LOG2E;
    }
    float m = pr[off + 32];
#pragma unroll
    for (int j = 1; j < 16; ++j) m = fmaxf(m, pr[off + 32 + j]);
    float s = 0.0f;
#pragma unroll
    for (int j = 0; j < 16; ++j) {
      float e = __builtin_amdgcn_exp2f((pr[off + 32 + j] - m) * LOG2E);
      Wt[l * 16 + j] = e;
      s += e;
    }
    float inv = __builtin_amdgcn_rcpf(s);
#pragma unroll
    for (int j = 0; j < 16; ++j) Wt[l * 16 + j] *= inv;
  }

  float uv = u[idx];
  float left = -1000.0f, right = 1000.0f;
#pragma unroll 1
  for (int it = 0; it < MAX_ITER; ++it) {
    float mid = 0.5f * (left + right);
    float x = mid;  // layer 0: real units; layers 1/2: log2 units
#pragma unroll
    for (int l = 0; l < 3; ++l) {
      float s = 0.0f;
#pragma unroll
      for (int j = 0; j < 16; ++j) {
        float t  = __builtin_fmaf(A[l * 16 + j], x, Bb[l * 16 + j]);
        float sg = __builtin_amdgcn_rcpf(1.0f + __builtin_amdgcn_exp2f(-t));
        s = __builtin_fmaf(Wt[l * 16 + j], sg, s);
      }
      if (l < 2) {
        float xp = __builtin_fmaf(s, 0.999999f, 5e-7f);  // s*(1-eps)+eps/2
        x = __builtin_amdgcn_logf(xp) - __builtin_amdgcn_logf(1.0f - xp);
      } else {
        x = s;
      }
    }
    float err = x - uv;
    left  = (err <= 0.0f) ? mid : left;
    right = (err >= 0.0f) ? mid : right;
  }
  out[idx] = 0.5f * (left + right);
}

// ---------------------------------------------------------------------------
extern "C" void kernel_launch(void* const* d_in, const int* in_sizes, int n_in,
                              void* d_out, int out_size, void* d_ws, size_t ws_size,
                              hipStream_t stream) {
  (void)in_sizes; (void)n_in; (void)out_size; (void)ws_size;
  const float* ctx = (const float*)d_in[0];
  const float* u   = (const float*)d_in[1];
  const float* W0  = (const float*)d_in[2];
  const float* b0  = (const float*)d_in[3];
  const float* W1  = (const float*)d_in[4];
  const float* b1  = (const float*)d_in[5];
  const float* W2  = (const float*)d_in[6];
  const float* b2  = (const float*)d_in[7];

  char* ws = (char*)d_ws;
  const size_t PARAM_BYTES = (size_t)BN_TOTAL * PDIM * sizeof(float);
  float* params = (float*)ws;
  unsigned short* w0f = (unsigned short*)(ws + PARAM_BYTES);
  unsigned short* w1f = w0f + (size_t)KT * NT0 * 32 * 16;
  unsigned short* w2f = w1f + (size_t)KT * NT0 * 32 * 16;

  // weight swizzle (L2-resident, trivial cost)
  prep_weights<<<(KT * NT0 * 32 + 255) / 256, 256, 0, stream>>>(W0, w0f, NT0, MLPD);
  prep_weights<<<(KT * NT0 * 32 + 255) / 256, 256, 0, stream>>>(W1, w1f, NT0, MLPD);
  prep_weights<<<(KT * NT2 * 32 + 255) / 256, 256, 0, stream>>>(W2, w2f, NT2, PDIM);

  // fused 3-layer WMMA MLP: 4 waves x 32 rows = 128 rows/block
  conditioner<<<BN_TOTAL / ROWS_PER_BLOCK, WAVES_PER_BLOCK * 32, 0, stream>>>(
      ctx, w0f, w1f, w2f, b0, b1, b2, params);

  // transcendental-bound bisection
  bisect<<<BN_TOTAL / 256, 256, 0, stream>>>(params, u, (float*)d_out);
}
